// HierarchicalDynamicFFN_52201032515630
// MI455X (gfx1250) — matrix operand order repaired and checked by reading, not compiled, not run on hardware
//
#include <hip/hip_runtime.h>

// ---------------------------------------------------------------------------
// Types / helpers
// ---------------------------------------------------------------------------
typedef __attribute__((ext_vector_type(16))) __bf16 v16bf;
typedef __attribute__((ext_vector_type(8)))  __bf16 v8bf;
typedef __attribute__((ext_vector_type(8)))  float  v8f;

__device__ __forceinline__ __bf16 f2bf(float f) {
  unsigned int u = __builtin_bit_cast(unsigned int, f);
  u += 0x7FFFu + ((u >> 16) & 1u);                 // round-to-nearest-even
  unsigned short h = (unsigned short)(u >> 16);
  return __builtin_bit_cast(__bf16, h);
}

__device__ __forceinline__ float gelu_exact(float v) {
  return 0.5f * v * (1.0f + erff(v * 0.7071067811865475f));
}

// Async 16B global -> LDS copy (CDNA5 VFLAT async op, tracked by ASYNCcnt).
// dst: LDS byte offset (generic LDS pointers carry it in the low 32 bits,
// ISA 10.2), src: 64-bit global address. GV mode (SADDR = off).
__device__ __forceinline__ void async_g2l_b128(unsigned lds_off, const void* gsrc) {
  asm volatile("global_load_async_to_lds_b128 %0, %1, off"
               :: "v"(lds_off), "v"((unsigned long long)gsrc) : "memory");
}
__device__ __forceinline__ void wait_async0() {
  asm volatile("s_wait_asynccnt 0x0" ::: "memory");
}

// ---------------------------------------------------------------------------
// Batched NT WMMA GEMM on bf16 operands:
//   D = act(alpha * A @ B^T + bias),  A:[M,K] bf16, B:[N,K] bf16 (row-major)
//   Optional f32 output C and/or bf16 output Cb (same layout/strides).
//   grid.z decomposed as (batch, head) via `heads`; per-axis element strides.
//
//   Block = 128 threads (4 waves) computing a 64x64 tile (wave = 16x64).
//   K-step 64: the 64x64 B tile shared by all 4 waves is staged into LDS via
//   double-buffered GLOBAL_LOAD_ASYNC_TO_LDS_B128 (prefetch of tile k+1
//   overlaps the 8 WMMAs on tile k -> one ASYNCcnt wait + one barrier per
//   64 K). Per 32-K half-step all 4 B fragments are preloaded from LDS
//   before the WMMA chain so dscnt is waited once per 4 matrix ops.
//   A fragments load directly (no intra-block reuse). All launch tilings are
//   exact (no partial tiles -> in-loop barriers are safe), K % 64 == 0.
// ---------------------------------------------------------------------------
__global__ void wmma_gemm(const __bf16* __restrict__ A, const __bf16* __restrict__ Bw,
                          const float* __restrict__ bias,
                          float* __restrict__ C, __bf16* __restrict__ Cb,
                          int M, int N, int K, int lda, int ldb, int ldc,
                          long aB, long aH, long bB, long bH, long cB, long cH,
                          int heads, int act, float alpha) {
  const int z = blockIdx.z;
  const int b = z / heads, h = z % heads;
  A  += (long)b * aB + (long)h * aH;
  Bw += (long)b * bB + (long)h * bH;

  const int lane = threadIdx.x & 31;
  const int wave = threadIdx.x >> 5;            // 0..3
  const int m0 = blockIdx.y * 64 + wave * 16;
  const int n0 = blockIdx.x * 64;

  const int row = lane & 15;                    // M row (A) / N col (B)
  const int sel = lane >> 4;                    // K-half select

  // ---- LDS staging of the 64x64 B tile, double buffered (2 x 8KB) ----
  __shared__ __bf16 lbs[2][64 * 64];
  const int tid   = threadIdx.x;                // 0..127
  const int srow  = tid >> 1;                   // B row this thread stages
  const int shalf = tid & 1;                    // which 32-element half
  const __bf16* bsrc = Bw + (long)(n0 + srow) * ldb + shalf * 32;
  __bf16* ld0 = &lbs[0][srow * 64 + shalf * 32];
  __bf16* ld1 = &lbs[1][srow * 64 + shalf * 32];

  auto stage = [&](const __bf16* g, __bf16* l) {
    const unsigned lo = (unsigned)(unsigned long long)(void*)l;
    async_g2l_b128(lo,       g);
    async_g2l_b128(lo + 16u, g + 8);
    async_g2l_b128(lo + 32u, g + 16);
    async_g2l_b128(lo + 48u, g + 24);
  };

  stage(bsrc, ld0);                             // tile 0 -> buf 0

  const __bf16* ap = A + (long)(m0 + row) * lda + sel * 16;
  v8f acc[4] = {};
  int buf = 0;

  for (int k0 = 0; k0 < K; k0 += 64) {
    wait_async0();                              // own copies of tile k0 done
    __syncthreads();                            // all waves' copies visible;
                                                // prior reads of other buf done
    if (k0 + 64 < K) stage(bsrc + k0 + 64, buf ? ld0 : ld1);

    const __bf16* lbb = &lbs[buf][row * 64 + sel * 16];
#pragma unroll
    for (int kk = 0; kk < 2; ++kk) {            // two 32-K half-steps
      v8bf a0 = *(const v8bf*)(ap + k0 + kk * 32);
      v8bf a1 = *(const v8bf*)(ap + k0 + kk * 32 + 8);
      v16bf af = __builtin_shufflevector(a0, a1, 0,1,2,3,4,5,6,7,8,9,10,11,12,13,14,15);

      const __bf16* lk = lbb + kk * 32;
      v16bf bf0 = *(const v16bf*)(lk + 0 * 16 * 64);   // preload all 4
      v16bf bf1 = *(const v16bf*)(lk + 1 * 16 * 64);   // B fragments,
      v16bf bf2 = *(const v16bf*)(lk + 2 * 16 * 64);   // then 4 wmma
      v16bf bf3 = *(const v16bf*)(lk + 3 * 16 * 64);   // back-to-back
      acc[0] = __builtin_amdgcn_wmma_f32_16x16x32_bf16(false, af, false, bf0, (short)0, acc[0], false, false);
      acc[1] = __builtin_amdgcn_wmma_f32_16x16x32_bf16(false, af, false, bf1, (short)0, acc[1], false, false);
      acc[2] = __builtin_amdgcn_wmma_f32_16x16x32_bf16(false, af, false, bf2, (short)0, acc[2], false, false);
      acc[3] = __builtin_amdgcn_wmma_f32_16x16x32_bf16(false, af, false, bf3, (short)0, acc[3], false, false);
    }
    buf ^= 1;
  }

  // ---- epilogue: alpha, bias, activation, f32 and/or bf16 store ----
  const long cbase = (long)b * cB + (long)h * cH;
#pragma unroll
  for (int t = 0; t < 4; ++t) {
    const int n = n0 + t * 16 + row;
    const float bv = bias ? bias[n] : 0.0f;
#pragma unroll
    for (int r = 0; r < 8; ++r) {
      const int m = m0 + r + sel * 8;
      float v = acc[t][r] * alpha + bv;
      if (act == 1) v = gelu_exact(v);
      const long o = cbase + (long)m * ldc + n;
      if (C)  C[o]  = v;
      if (Cb) Cb[o] = f2bf(v);
    }
  }
}

// ---------------------------------------------------------------------------
// Elementwise f32 -> bf16
// ---------------------------------------------------------------------------
__global__ void cvt_f32_bf16(const float* __restrict__ in, __bf16* __restrict__ out, long n) {
  const long g = (long)blockIdx.x * blockDim.x + threadIdx.x;
  if (g < n) out[g] = f2bf(in[g]);
}

// ---------------------------------------------------------------------------
// Batched bf16 transpose: out[c, r] = in[r, c]   (in: [R,C], out: [C,R])
// ---------------------------------------------------------------------------
__global__ void bf_transpose(const __bf16* __restrict__ in, __bf16* __restrict__ out,
                             int R, int C, int ldin, int ldout,
                             long iB, long iH, long oB, long oH, int heads) {
  const int z = blockIdx.z;
  const int b = z / heads, h = z % heads;
  in  += (long)b * iB + (long)h * iH;
  out += (long)b * oB + (long)h * oH;
  const long g = (long)blockIdx.x * blockDim.x + threadIdx.x;   // over C*R
  if (g >= (long)C * R) return;
  const int r = (int)(g % R);
  const int c = (int)(g / R);
  out[(long)c * ldout + r] = in[(long)r * ldin + c];
}

// ---------------------------------------------------------------------------
// Row softmax: read f32 scores, write normalized bf16. One block per row.
// L <= 1024 (4 elements/thread cached in registers).
// ---------------------------------------------------------------------------
__global__ void softmax_rows_bf16(const float* __restrict__ x, __bf16* __restrict__ y, int L) {
  const float* row = x + (long)blockIdx.x * L;
  __bf16* yr = y + (long)blockIdx.x * L;
  __shared__ float red[256];
  const int tid = threadIdx.x;
  float mx = -3.4e38f;
  for (int i = tid; i < L; i += 256) mx = fmaxf(mx, row[i]);
  red[tid] = mx; __syncthreads();
  for (int s = 128; s > 0; s >>= 1) { if (tid < s) red[tid] = fmaxf(red[tid], red[tid + s]); __syncthreads(); }
  mx = red[0]; __syncthreads();
  float ev[4];
  float sum = 0.0f;
  int c = 0;
  for (int i = tid; i < L; i += 256) { float e = expf(row[i] - mx); ev[c++] = e; sum += e; }
  red[tid] = sum; __syncthreads();
  for (int s = 128; s > 0; s >>= 1) { if (tid < s) red[tid] += red[tid + s]; __syncthreads(); }
  const float inv = 1.0f / red[0];
  c = 0;
  for (int i = tid; i < L; i += 256) yr[i] = f2bf(ev[c++] * inv);
}

// ---------------------------------------------------------------------------
// out = LayerNorm(a + b) * g + beta. One block per row of length L.
// ---------------------------------------------------------------------------
__global__ void add_layernorm(const float* __restrict__ a, const float* __restrict__ bsrc,
                              const float* __restrict__ g, const float* __restrict__ beta,
                              float* __restrict__ out, int L) {
  const long r = blockIdx.x;
  const float* pa = a + r * L;
  const float* pb = bsrc + r * L;
  __shared__ float red[256];
  const int tid = threadIdx.x;
  float s = 0.0f;
  for (int i = tid; i < L; i += 256) s += pa[i] + pb[i];
  red[tid] = s; __syncthreads();
  for (int k = 128; k > 0; k >>= 1) { if (tid < k) red[tid] += red[tid + k]; __syncthreads(); }
  const float mu = red[0] / (float)L; __syncthreads();
  float v = 0.0f;
  for (int i = tid; i < L; i += 256) { float d = pa[i] + pb[i] - mu; v += d * d; }
  red[tid] = v; __syncthreads();
  for (int k = 128; k > 0; k >>= 1) { if (tid < k) red[tid] += red[tid + k]; __syncthreads(); }
  const float inv = rsqrtf(red[0] / (float)L + 1e-5f);
  for (int i = tid; i < L; i += 256)
    out[r * L + i] = (pa[i] + pb[i] - mu) * inv * g[i] + beta[i];
}

// ---------------------------------------------------------------------------
// Column stats over S rows per batch. x: [B*S, n]. mode 0: 0.7*mean+0.3*max
// (router final_scores, using pos_imp==1/S identity). mode 1: mean.
// ---------------------------------------------------------------------------
__global__ void col_stats(const float* __restrict__ x, int S, int n,
                          float* __restrict__ out, int mode) {
  const int g = blockIdx.x * blockDim.x + threadIdx.x;   // b*n + c
  const int b = g / n, c = g % n;
  const float* p = x + (long)b * S * n + c;
  float sum = 0.0f, mx = -3.4e38f;
  for (int s = 0; s < S; ++s) { float v = p[(long)s * n]; sum += v; mx = fmaxf(mx, v); }
  const float mean = sum / (float)S;
  out[g] = (mode == 0) ? (0.7f * mean + 0.3f * mx) : mean;
}

// ---------------------------------------------------------------------------
// Deterministic rank-based top-k: out[b*k + rank] = index. n <= 2048.
// ---------------------------------------------------------------------------
__global__ void topk_idx(const float* __restrict__ scores, int n, int k,
                         int* __restrict__ out) {
  __shared__ float s[2048];
  const float* p = scores + (long)blockIdx.x * n;
  for (int i = threadIdx.x; i < n; i += blockDim.x) s[i] = p[i];
  __syncthreads();
  for (int i = threadIdx.x; i < n; i += blockDim.x) {
    const float v = s[i];
    int rank = 0;
    for (int j = 0; j < n; ++j) rank += (s[j] > v) || (s[j] == v && j < i);
    if (rank < k) out[blockIdx.x * k + rank] = i;
  }
}

// ---------------------------------------------------------------------------
// Gathers (routing_weights == one_hot in forward => gather == select cols).
// ---------------------------------------------------------------------------
__global__ void gather_cols_bf16(const float* __restrict__ in, const int* __restrict__ idx,
                                 __bf16* __restrict__ out, int rowsPerB, int nin, int k,
                                 int sharedRows) {
  const long g = (long)blockIdx.x * blockDim.x + threadIdx.x;  // [B, rowsPerB, k]
  const int  j = (int)(g % k);
  const long r = g / k;
  const int  b = (int)(r / rowsPerB);
  const long inRow = sharedRows ? (r % rowsPerB) : r;
  out[g] = f2bf(in[inRow * nin + idx[b * k + j]]);
}

// selOT[b, c, j] = proc_out[idx[b*k+j], c]  -> B operand already transposed
__global__ void gather_rows_t_bf16(const float* __restrict__ in, const int* __restrict__ idx,
                                   __bf16* __restrict__ out, int k, int d) {
  const long g = (long)blockIdx.x * blockDim.x + threadIdx.x;  // [B, d, k]
  const int  j = (int)(g % k);
  const long t = g / k;
  const int  c = (int)(t % d);
  const int  b = (int)(t / d);
  out[g] = f2bf(in[(long)idx[b * k + j] * d + c]);
}

// ---------------------------------------------------------------------------
// Launch
// ---------------------------------------------------------------------------
extern "C" void kernel_launch(void* const* d_in, const int* in_sizes, int n_in,
                              void* d_out, int out_size, void* d_ws, size_t ws_size,
                              hipStream_t stream) {
  const float* x        = (const float*)d_in[0];
  const float* r_in_w   = (const float*)d_in[3];
  const float* r_in_b   = (const float*)d_in[4];
  const float* r_out_w  = (const float*)d_in[5];
  const float* r_out_b  = (const float*)d_in[6];
  const float* c2n_w    = (const float*)d_in[7];
  const float* c2n_b    = (const float*)d_in[8];
  const float* patterns = (const float*)d_in[9];
  const float* n_in_w   = (const float*)d_in[10];
  const float* n_in_b   = (const float*)d_in[11];
  const float* n_out_w  = (const float*)d_in[12];
  const float* n_out_b  = (const float*)d_in[13];
  const float* ln_g     = (const float*)d_in[14];
  const float* ln_b     = (const float*)d_in[15];
  const float* proc_w   = (const float*)d_in[16];
  const float* proc_out = (const float*)d_in[17];
  float* out = (float*)d_out;

  const int Bb = 8, S = 1024, D = 1024, NI = 512, NP = 2048;
  const int HR = 8, HN = 4, DH = 128, KIN = 256, KPR = 1024;
  const long M = (long)Bb * S;                       // 8192 tokens
  const float iscale = 0.08838834764831845f;         // 1/sqrt(128)

  // ---- workspace layout ----
  // f32 region
  float* sc   = (float*)d_ws;                        // [64,S,S] scores (reused [32,S,S], [B,S,NP])
  float* aff  = sc   + 64L * S * S;                  // [8192,512]
  float* loc  = aff  + M * NI;                       // local_acts f32
  float* ao2  = loc  + M * NI;                       // neuron attn out f32
  float* ina  = ao2  + M * NI;                       // input_acts f32
  float* fsc  = ina  + M * NI;                       // [8,2048]
  float* psc  = fsc  + (long)Bb * NP;                // [8,2048]
  int*   idxIn= (int*)(psc + (long)Bb * NP);         // [8,256]
  int*   idxPr= idxIn + Bb * KIN;                    // [8,1024]
  // bf16 region
  __bf16* xb    = (__bf16*)(idxPr + Bb * KPR);
  __bf16* wrin  = xb    + M * D;                     // [3072,1024]
  __bf16* wrout = wrin  + 3072L * D;                 // [1024,1024]
  __bf16* wc2n  = wrout + 1024L * D;                 // [512,1024]
  __bf16* wpat  = wc2n  + 512L * D;                  // [512,1024]
  __bf16* wnin  = wpat  + 512L * D;                  // [1536,512]
  __bf16* wnout = wnin  + 1536L * NI;                // [512,512]
  __bf16* qkvb  = wnout + 512L * NI;                 // [8192,3072] (reused [8192,1536])
  __bf16* vT    = qkvb  + M * 3072;                  // [64,128,1024] (reused [32,128,1024])
  __bf16* attb  = vT    + 64L * DH * S;              // [64,S,S] bf16 attn (reused [32,S,S])
  __bf16* ctxb  = attb  + 64L * S * S;               // [8192,1024] (reused [8192,512])
  __bf16* attdb = ctxb  + M * D;                     // attended bf16 [8192,1024]
  __bf16* locb  = attdb + M * D;                     // [8192,512]
  __bf16* selA  = locb  + M * NI;                    // [8,1024,256]
  __bf16* selW  = selA  + (long)Bb * S * KIN;        // [8,2048,256]
  __bf16* selPA = selW  + (long)Bb * NP * KIN;       // [8,1024,1024]
  __bf16* selOT = selPA + (long)Bb * S * KPR;        // [8,1024,1024]

  const dim3 blk(128), b256(256);

  // 0. stage bf16 operands
  cvt_f32_bf16<<<dim3((unsigned)(M * D / 256)),      b256, 0, stream>>>(x,        xb,    M * D);
  cvt_f32_bf16<<<dim3((unsigned)(3072L * D / 256)),  b256, 0, stream>>>(r_in_w,   wrin,  3072L * D);
  cvt_f32_bf16<<<dim3((unsigned)(1024L * D / 256)),  b256, 0, stream>>>(r_out_w,  wrout, 1024L * D);
  cvt_f32_bf16<<<dim3((unsigned)(512L * D / 256)),   b256, 0, stream>>>(c2n_w,    wc2n,  512L * D);
  cvt_f32_bf16<<<dim3((unsigned)(512L * D / 256)),   b256, 0, stream>>>(patterns, wpat,  512L * D);
  cvt_f32_bf16<<<dim3((unsigned)(1536L * NI / 256)), b256, 0, stream>>>(n_in_w,   wnin,  1536L * NI);
  cvt_f32_bf16<<<dim3((unsigned)(512L * NI / 256)),  b256, 0, stream>>>(n_out_w,  wnout, 512L * NI);

  // 1. qkv = x @ r_in_w^T + r_in_b  -> bf16 only
  wmma_gemm<<<dim3(3072/64, 8192/64, 1), blk, 0, stream>>>(
      xb, wrin, r_in_b, nullptr, qkvb, 8192, 3072, 1024, 1024, 1024, 3072,
      0,0, 0,0, 0,0, 1, 0, 1.0f);
  // 2. V^T per (b,h): [S,Dh] -> [Dh,S]
  bf_transpose<<<dim3((unsigned)(DH * S / 256), 1, Bb*HR), b256, 0, stream>>>(
      qkvb + 2048, vT, S, DH, 3072, S,
      (long)S*3072, 128, (long)HR*DH*S, (long)DH*S, HR);
  // 3. scores = (Q @ K^T) * iscale per (b,h) -> f32
  wmma_gemm<<<dim3(S/64, S/64, Bb*HR), blk, 0, stream>>>(
      qkvb, qkvb + 1024, nullptr, sc, nullptr, S, S, DH, 3072, 3072, S,
      (long)S*3072, 128, (long)S*3072, 128, (long)HR*S*S, (long)S*S, HR, 0, iscale);
  // 4. softmax -> bf16 attn
  softmax_rows_bf16<<<dim3(Bb*HR*S), b256, 0, stream>>>(sc, attb, S);
  // 5. ctx = attn @ (V^T)^T -> interleaved heads, bf16 only
  wmma_gemm<<<dim3(DH/64, S/64, Bb*HR), blk, 0, stream>>>(
      attb, vT, nullptr, nullptr, ctxb, S, DH, S, S, S, 1024,
      (long)HR*S*S, (long)S*S, (long)HR*DH*S, (long)DH*S, (long)S*1024, 128, HR, 0, 1.0f);
  // 6. attended = ctx @ r_out_w^T + r_out_b -> bf16 only
  wmma_gemm<<<dim3(1024/64, 8192/64, 1), blk, 0, stream>>>(
      ctxb, wrout, r_out_b, nullptr, attdb, 8192, 1024, 1024, 1024, 1024, 1024,
      0,0, 0,0, 0,0, 1, 0, 1.0f);
  // 7. neuron_affinity -> f32
  wmma_gemm<<<dim3(512/64, 8192/64, 1), blk, 0, stream>>>(
      attdb, wc2n, c2n_b, aff, nullptr, 8192, 512, 1024, 1024, 1024, 512,
      0,0, 0,0, 0,0, 1, 0, 1.0f);
  // 8. final_scores = 0.7*mean + 0.3*max ; top-k input neurons
  col_stats<<<dim3(Bb*NI/256), b256, 0, stream>>>(aff, S, NI, fsc, 0);
  topk_idx<<<dim3(Bb), b256, 0, stream>>>(fsc, NI, KIN, idxIn);
  // 9. local = gelu(attended @ patterns^T) -> f32 + bf16
  wmma_gemm<<<dim3(512/64, 8192/64, 1), blk, 0, stream>>>(
      attdb, wpat, nullptr, loc, locb, 8192, 512, 1024, 1024, 1024, 512,
      0,0, 0,0, 0,0, 1, 1, 1.0f);
  // 10. qkv2 = local @ n_in_w^T + n_in_b -> bf16 (reuses qkvb)
  wmma_gemm<<<dim3(1536/64, 8192/64, 1), blk, 0, stream>>>(
      locb, wnin, n_in_b, nullptr, qkvb, 8192, 1536, 512, 512, 512, 1536,
      0,0, 0,0, 0,0, 1, 0, 1.0f);
  // 11. V2^T per (b,h)
  bf_transpose<<<dim3((unsigned)(DH * S / 256), 1, Bb*HN), b256, 0, stream>>>(
      qkvb + 1024, vT, S, DH, 1536, S,
      (long)S*1536, 128, (long)HN*DH*S, (long)DH*S, HN);
  // 12. scores2 -> f32 (reuses sc)
  wmma_gemm<<<dim3(S/64, S/64, Bb*HN), blk, 0, stream>>>(
      qkvb, qkvb + 512, nullptr, sc, nullptr, S, S, DH, 1536, 1536, S,
      (long)S*1536, 128, (long)S*1536, 128, (long)HN*S*S, (long)S*S, HN, 0, iscale);
  // 13. softmax -> bf16
  softmax_rows_bf16<<<dim3(Bb*HN*S), b256, 0, stream>>>(sc, attb, S);
  // 14. ctx2 -> bf16 (reuses ctxb, [8192,512])
  wmma_gemm<<<dim3(DH/64, S/64, Bb*HN), blk, 0, stream>>>(
      attb, vT, nullptr, nullptr, ctxb, S, DH, S, S, S, 512,
      (long)HN*S*S, (long)S*S, (long)HN*DH*S, (long)DH*S, (long)S*512, 128, HN, 0, 1.0f);
  // 15. attn_out2 -> f32
  wmma_gemm<<<dim3(512/64, 8192/64, 1), blk, 0, stream>>>(
      ctxb, wnout, n_out_b, ao2, nullptr, 8192, 512, 512, 512, 512, 512,
      0,0, 0,0, 0,0, 1, 0, 1.0f);
  // 16. input_acts = LN(local + attn_out2)
  add_layernorm<<<dim3(8192), b256, 0, stream>>>(loc, ao2, ln_g, ln_b, ina, NI);
  // 17. gathers for process stage
  gather_cols_bf16<<<dim3((unsigned)((long)Bb*S*KIN/256)), b256, 0, stream>>>(
      ina, idxIn, selA, S, NI, KIN, 0);
  gather_cols_bf16<<<dim3((unsigned)((long)Bb*NP*KIN/256)), b256, 0, stream>>>(
      proc_w, idxIn, selW, NP, NI, KIN, 1);
  // 18. process_acts = gelu(selA @ selW^T) per batch -> f32 (reuses sc)
  wmma_gemm<<<dim3(NP/64, S/64, Bb), blk, 0, stream>>>(
      selA, selW, nullptr, sc, nullptr, S, NP, KIN, KIN, KIN, NP,
      (long)S*KIN, 0, (long)NP*KIN, 0, (long)S*NP, 0, 1, 1, 1.0f);
  // 19. process_scores = mean_S ; top-k process neurons
  col_stats<<<dim3(Bb*NP/256), b256, 0, stream>>>(sc, S, NP, psc, 1);
  topk_idx<<<dim3(Bb), b256, 0, stream>>>(psc, NP, KPR, idxPr);
  // 20. gather selected activations / transposed output rows
  gather_cols_bf16<<<dim3((unsigned)((long)Bb*S*KPR/256)), b256, 0, stream>>>(
      sc, idxPr, selPA, S, NP, KPR, 0);
  gather_rows_t_bf16<<<dim3((unsigned)((long)Bb*D*KPR/256)), b256, 0, stream>>>(
      proc_out, idxPr, selOT, KPR, D);
  // 21. output = selPA @ selOT^T per batch -> f32 out
  wmma_gemm<<<dim3(D/64, S/64, Bb), blk, 0, stream>>>(
      selPA, selOT, nullptr, out, nullptr, S, D, KPR, KPR, KPR, D,
      (long)S*KPR, 0, (long)D*KPR, 0, (long)S*D, 0, 1, 0, 1.0f);
}
